// LinearTransformerBlock_52707838656612
// MI455X (gfx1250) — matrix operand order repaired and checked by reading, not compiled, not run on hardware
//
#include <hip/hip_runtime.h>
#include <hip/hip_bf16.h>
#include <math.h>

typedef _Float16 half_t;
typedef __attribute__((ext_vector_type(16))) _Float16 v16h;
typedef __attribute__((ext_vector_type(8)))  _Float16 v8h;
typedef __attribute__((ext_vector_type(8)))  float    v8f;

#define T_SEQ   2048
#define D_MODEL 1024
#define N_HEADS 16
#define D_HEAD  64
#define D_FF    4096

#define BM 128
#define BN 128
#define BK 32
#define GEMM_THREADS 512   // 16 waves: 4x4 grid of 32x32 wave tiles

// ---------------------------------------------------------------------------
// WMMA fragment helpers (wave32, V_WMMA_F32_16X16X32_F16)
// A (16x32 f16, row-major source):
//   lanes 0-15 : row M=lane,    halves = K 0..7  and K 16..23
//   lanes 16-31: row M=lane-16, halves = K 8..15 and K 24..31
// B loaded from B^T (N-major, K contiguous):
//   lanes 0-15 : col N=lane,    halves = K 0..7  and K 16..23
//   lanes 16-31: col N=lane-16, halves = K 8..15 and K 24..31
// C/D f32: VGPR r -> row m0 + (lane>=16 ? 8 : 0) + r, col n0 + (lane&15)
// ---------------------------------------------------------------------------
__device__ __forceinline__ v16h load_a_frag(const half_t* A, int lda, int lane) {
  const int row  = lane & 15;
  const int koff = (lane >> 4) << 3;
  const half_t* p = A + row * lda + koff;
  v8h lo = *(const v8h*)(p);
  v8h hi = *(const v8h*)(p + 16);
  v16h a;
#pragma unroll
  for (int i = 0; i < 8; ++i) { a[i] = lo[i]; a[i + 8] = hi[i]; }
  return a;
}

__device__ __forceinline__ v16h load_bt_frag(const half_t* Bt, int ldb, int lane) {
  const int col = lane & 15;
  const int kb  = (lane >> 4) << 3;
  const half_t* p = Bt + col * ldb + kb;
  v8h b0 = *(const v8h*)(p);
  v8h b1 = *(const v8h*)(p + 16);
  v16h b;
#pragma unroll
  for (int i = 0; i < 8; ++i) { b[i] = b0[i]; b[i + 8] = b1[i]; }
  return b;
}

__device__ __forceinline__ v8f wmma_f16(v16h a, v16h b, v8f c) {
  return __builtin_amdgcn_wmma_f32_16x16x32_f16(false, a, false, b, (short)0, c,
                                                false, false);
}

// ---------------------------------------------------------------------------
// Async global->LDS copy (gfx1250): 16 bytes per lane, tracked by ASYNCcnt.
// LDS address = low 32 bits of the generic address of a __shared__ pointer
// (ISA 10.2, LDS aperture).
// ---------------------------------------------------------------------------
__device__ __forceinline__ unsigned lds_addr(const void* p) {
  return (unsigned)(unsigned long long)p;
}

__device__ __forceinline__ void async_cp16(unsigned lds, const half_t* g) {
  asm volatile("global_load_async_to_lds_b128 %0, %1, off"
               :: "v"(lds), "v"(g) : "memory");
}

__device__ __forceinline__ void wait_async0() {
  asm volatile("s_wait_asynccnt 0" ::: "memory");
}

// ---------------------------------------------------------------------------
// 128x128 block tile, LDS double-buffered over K (BK=32), 512 threads.
// Each thread stages one 16B A-segment + one 16B B-segment per K-step
// (addresses hoisted; global pointers bumped by BK per stage).
// Each wave computes a 32x32 sub-tile = 4 WMMA accumulators.
// Epilogue mapping: accMN[r] -> row erow + M*16 + r, col ecol + N*16
// ---------------------------------------------------------------------------
#define STAGED_GEMM(Aptr, lda, Btp, ldb, KDIM)                                  \
  __shared__ __align__(16) half_t shA[2][BM * BK];                              \
  __shared__ __align__(16) half_t shB[2][BN * BK];                              \
  const int tid  = threadIdx.x;                                                 \
  const int wave = tid >> 5;                                                    \
  const int lane = tid & 31;                                                    \
  const int wm   = (wave >> 2) * 32;                                            \
  const int wn   = (wave & 3) * 32;                                             \
  const int srow = tid >> 2;                                                    \
  const int ssg  = (tid & 3) * 8;                                               \
  const half_t* gAp = (Aptr) + ((size_t)blockIdx.x * BM + srow) * (lda) + ssg;  \
  const half_t* gBp = (Btp) + ((size_t)blockIdx.y * BN + srow) * (ldb) + ssg;   \
  const unsigned lA0 = lds_addr(&shA[0][srow * BK + ssg]);                      \
  const unsigned lA1 = lds_addr(&shA[1][srow * BK + ssg]);                      \
  const unsigned lB0 = lds_addr(&shB[0][srow * BK + ssg]);                      \
  const unsigned lB1 = lds_addr(&shB[1][srow * BK + ssg]);                      \
  v8f acc00 = {}, acc01 = {}, acc10 = {}, acc11 = {};                           \
  async_cp16(lA0, gAp);                                                         \
  async_cp16(lB0, gBp);                                                         \
  gAp += BK; gBp += BK;                                                         \
  const int nk = (KDIM) / BK;                                                   \
  for (int kt = 0; kt < nk; ++kt) {                                             \
    wait_async0();                                                              \
    __syncthreads();                                                            \
    if (kt + 1 < nk) {                                                          \
      if (kt & 1) { async_cp16(lA0, gAp); async_cp16(lB0, gBp); }               \
      else        { async_cp16(lA1, gAp); async_cp16(lB1, gBp); }               \
      gAp += BK; gBp += BK;                                                     \
    }                                                                           \
    const half_t* Ab = shA[kt & 1];                                             \
    const half_t* Bb = shB[kt & 1];                                             \
    v16h af0 = load_a_frag(Ab + (wm) * BK, BK, lane);                           \
    v16h af1 = load_a_frag(Ab + (wm + 16) * BK, BK, lane);                      \
    v16h bf0 = load_bt_frag(Bb + (wn) * BK, BK, lane);                          \
    v16h bf1 = load_bt_frag(Bb + (wn + 16) * BK, BK, lane);                     \
    acc00 = wmma_f16(af0, bf0, acc00);                                          \
    acc01 = wmma_f16(af0, bf1, acc01);                                          \
    acc10 = wmma_f16(af1, bf0, acc10);                                          \
    acc11 = wmma_f16(af1, bf1, acc11);                                          \
    __syncthreads();                                                            \
  }                                                                             \
  const int erow = blockIdx.x * BM + wm + ((lane >> 4) << 3);                   \
  const int ecol = blockIdx.y * BN + wn + (lane & 15);

// ---------------------------------------------------------------------------
// Weight convert: W (Kd x Nd f32, row-major) -> W^T (Nd x Kd f16)
// ---------------------------------------------------------------------------
__global__ void cvt_t_kernel(const float* __restrict__ W, half_t* __restrict__ Wt,
                             int Kd, int Nd) {
  int idx = blockIdx.x * 256 + threadIdx.x;
  if (idx >= Kd * Nd) return;
  int n = idx / Kd;
  int k = idx - n * Kd;
  Wt[idx] = (half_t)W[(size_t)k * Nd + n];
}

// ---------------------------------------------------------------------------
// LayerNorm row kernel: f32 in -> f16 out (one block per row)
// ---------------------------------------------------------------------------
__global__ void ln_kernel(const float* __restrict__ X, const float* __restrict__ g,
                          const float* __restrict__ b, half_t* __restrict__ Y) {
  __shared__ float red[256];
  const int row = blockIdx.x;
  const int tid = threadIdx.x;
  const float* xr = X + (size_t)row * D_MODEL;
  float s = 0.f;
  for (int i = tid; i < D_MODEL; i += 256) s += xr[i];
  red[tid] = s;
  __syncthreads();
  for (int o = 128; o > 0; o >>= 1) {
    if (tid < o) red[tid] += red[tid + o];
    __syncthreads();
  }
  const float mean = red[0] * (1.0f / D_MODEL);
  __syncthreads();
  float v = 0.f;
  for (int i = tid; i < D_MODEL; i += 256) {
    float d = xr[i] - mean;
    v += d * d;
  }
  red[tid] = v;
  __syncthreads();
  for (int o = 128; o > 0; o >>= 1) {
    if (tid < o) red[tid] += red[tid + o];
    __syncthreads();
  }
  const float rstd = rsqrtf(red[0] * (1.0f / D_MODEL) + 1e-5f);
  for (int i = tid; i < D_MODEL; i += 256)
    Y[(size_t)row * D_MODEL + i] = (half_t)((xr[i] - mean) * rstd * g[i] + b[i]);
}

// ---------------------------------------------------------------------------
// Q = phi(Xn @ Wq): stored [h][t][dh]
// ---------------------------------------------------------------------------
__global__ void gemm_q_phi(const half_t* __restrict__ Xn, const half_t* __restrict__ Wt,
                           half_t* __restrict__ Qn) {
  STAGED_GEMM(Xn, D_MODEL, Wt, D_MODEL, D_MODEL)
  auto epi = [&](v8f cf, int mi, int ni) {
    const int colg = ecol + ni * 16;
    const int h = colg >> 6, dh = colg & 63;
#pragma unroll
    for (int r = 0; r < 8; ++r) {
      const int t = erow + mi * 16 + r;
      float q = cf[r];
      q = q > 0.f ? q + 1.f : expf(q);
      Qn[((size_t)h * T_SEQ + t) * D_HEAD + dh] = (half_t)q;
    }
  };
  epi(acc00, 0, 0); epi(acc01, 0, 1); epi(acc10, 1, 0); epi(acc11, 1, 1);
}

// K = phi(Xn @ Wk): stored both [h][t][dh] (Kn) and [h][dh][t] (Kt)
__global__ void gemm_k_phi(const half_t* __restrict__ Xn, const half_t* __restrict__ Wt,
                           half_t* __restrict__ Kn, half_t* __restrict__ Kt) {
  STAGED_GEMM(Xn, D_MODEL, Wt, D_MODEL, D_MODEL)
  auto epi = [&](v8f cf, int mi, int ni) {
    const int colg = ecol + ni * 16;
    const int h = colg >> 6, dh = colg & 63;
#pragma unroll
    for (int r = 0; r < 8; ++r) {
      const int t = erow + mi * 16 + r;
      float kv = cf[r];
      kv = kv > 0.f ? kv + 1.f : expf(kv);
      Kn[((size_t)h * T_SEQ + t) * D_HEAD + dh] = (half_t)kv;
      Kt[((size_t)h * D_HEAD + dh) * T_SEQ + t] = (half_t)kv;
    }
  };
  epi(acc00, 0, 0); epi(acc01, 0, 1); epi(acc10, 1, 0); epi(acc11, 1, 1);
}

// V = Xn @ Wv: stored transposed [h][dv][t]
__global__ void gemm_v_t(const half_t* __restrict__ Xn, const half_t* __restrict__ Wt,
                         half_t* __restrict__ Vt) {
  STAGED_GEMM(Xn, D_MODEL, Wt, D_MODEL, D_MODEL)
  auto epi = [&](v8f cf, int mi, int ni) {
    const int colg = ecol + ni * 16;
    const int h = colg >> 6, dv = colg & 63;
#pragma unroll
    for (int r = 0; r < 8; ++r) {
      const int t = erow + mi * 16 + r;
      Vt[((size_t)h * D_HEAD + dv) * T_SEQ + t] = (half_t)cf[r];
    }
  };
  epi(acc00, 0, 0); epi(acc01, 0, 1); epi(acc10, 1, 0); epi(acc11, 1, 1);
}

// X1 = X + Attn @ Wo   (f32 out)
__global__ void gemm_wo_res(const half_t* __restrict__ A, const half_t* __restrict__ Wt,
                            const float* __restrict__ X, float* __restrict__ X1) {
  STAGED_GEMM(A, D_MODEL, Wt, D_MODEL, D_MODEL)
  auto epi = [&](v8f cf, int mi, int ni) {
    const int colg = ecol + ni * 16;
#pragma unroll
    for (int r = 0; r < 8; ++r) {
      const size_t t = (size_t)(erow + mi * 16 + r) * D_MODEL;
      X1[t + colg] = X[t + colg] + cf[r];
    }
  };
  epi(acc00, 0, 0); epi(acc01, 0, 1); epi(acc10, 1, 0); epi(acc11, 1, 1);
}

// G = gelu(H @ Wf1)  (f16 out, exact erf GELU)
__global__ void gemm_ffn1_gelu(const half_t* __restrict__ A, const half_t* __restrict__ Wt,
                               half_t* __restrict__ G) {
  STAGED_GEMM(A, D_MODEL, Wt, D_MODEL, D_MODEL)
  auto epi = [&](v8f cf, int mi, int ni) {
    const int colg = ecol + ni * 16;
#pragma unroll
    for (int r = 0; r < 8; ++r) {
      const int t = erow + mi * 16 + r;
      float xv = cf[r];
      xv = 0.5f * xv * (1.f + erff(xv * 0.70710678f));
      G[(size_t)t * D_FF + colg] = (half_t)xv;
    }
  };
  epi(acc00, 0, 0); epi(acc01, 0, 1); epi(acc10, 1, 0); epi(acc11, 1, 1);
}

// Out = X1 + G @ Wf2  (f32 out)
__global__ void gemm_ffn2_res(const half_t* __restrict__ A, const half_t* __restrict__ Wt,
                              const float* __restrict__ X1, float* __restrict__ Out) {
  STAGED_GEMM(A, D_FF, Wt, D_FF, D_FF)
  auto epi = [&](v8f cf, int mi, int ni) {
    const int colg = ecol + ni * 16;
#pragma unroll
    for (int r = 0; r < 8; ++r) {
      const size_t t = (size_t)(erow + mi * 16 + r) * D_MODEL;
      Out[t + colg] = X1[t + colg] + cf[r];
    }
  };
  epi(acc00, 0, 0); epi(acc01, 0, 1); epi(acc10, 1, 0); epi(acc11, 1, 1);
}

// ---------------------------------------------------------------------------
// Chunked causal linear attention. One workgroup (8 waves) per head; chunk=64.
// S (64x64 f32) and z (64 f32) carried in LDS across chunks.
//   num = Qc @ S_prev + tril(Qc Kc^T) @ Vc
//   den = max(Qc . z_prev + rowsum(tril(Qc Kc^T)), 1e-6)
//   S  += Kc^T @ Vc ;  z += colsum(Kc)
// ---------------------------------------------------------------------------
__global__ void attn_kernel(const half_t* __restrict__ Qn, const half_t* __restrict__ Kn,
                            const half_t* __restrict__ Kt, const half_t* __restrict__ Vt,
                            half_t* __restrict__ Out) {
  __shared__ __align__(16) float  Ssh[D_HEAD * D_HEAD];   // S, dh-major
  __shared__ __align__(16) half_t Sth[D_HEAD * D_HEAD];   // S^T f16 (dv-major)
  __shared__ __align__(16) half_t A2m[64 * 64];           // masked Q K^T, f16
  __shared__ float den[64];
  __shared__ float zsh[D_HEAD];

  const int h    = blockIdx.x;
  const int tid  = threadIdx.x;
  const int wave = tid >> 5;
  const int lane = tid & 31;

  const half_t* Qh  = Qn + (size_t)h * T_SEQ * D_HEAD;
  const half_t* Knh = Kn + (size_t)h * T_SEQ * D_HEAD;
  const half_t* Kth = Kt + (size_t)h * D_HEAD * T_SEQ;
  const half_t* Vth = Vt + (size_t)h * D_HEAD * T_SEQ;

  for (int i = tid; i < D_HEAD * D_HEAD; i += 256) {
    Ssh[i] = 0.f;
    Sth[i] = (half_t)0.f;
  }
  if (tid < D_HEAD) zsh[tid] = 0.f;
  __syncthreads();

  const int m0    = (wave >> 1) * 16;
  const int ncp   = (wave & 1) * 32;
  const int crow  = m0 + ((lane >> 4) << 3);
  const int ccol0 = ncp + (lane & 15);
  const int ccol1 = ccol0 + 16;

  for (int ch = 0; ch < T_SEQ / 64; ++ch) {
    const int t0 = ch * 64;
    if (tid < 64) den[tid] = 0.f;
    __syncthreads();

    // num_inter = Qc @ S_prev   (B^T = Sth)
    v8f acc0 = {}, acc1 = {};
    {
      const half_t* Ap = Qh + (size_t)(t0 + m0) * D_HEAD;
#pragma unroll
      for (int k = 0; k < D_HEAD; k += 32) {
        v16h a  = load_a_frag(Ap + k, D_HEAD, lane);
        v16h b0 = load_bt_frag(Sth + ncp * D_HEAD + k, D_HEAD, lane);
        v16h b1 = load_bt_frag(Sth + (ncp + 16) * D_HEAD + k, D_HEAD, lane);
        acc0 = wmma_f16(a, b0, acc0);
        acc1 = wmma_f16(a, b1, acc1);
      }
    }
    // A = Qc @ Kc^T, causal mask, intra-chunk den rowsums, stash f16
    {
      v8f s0 = {}, s1 = {};
      const half_t* Ap = Qh + (size_t)(t0 + m0) * D_HEAD;
      const half_t* Bp = Knh + (size_t)t0 * D_HEAD;   // (Kc^T)^T = Kc, t-major
#pragma unroll
      for (int k = 0; k < D_HEAD; k += 32) {
        v16h a  = load_a_frag(Ap + k, D_HEAD, lane);
        v16h b0 = load_bt_frag(Bp + ncp * D_HEAD + k, D_HEAD, lane);
        v16h b1 = load_bt_frag(Bp + (ncp + 16) * D_HEAD + k, D_HEAD, lane);
        s0 = wmma_f16(a, b0, s0);
        s1 = wmma_f16(a, b1, s1);
      }
#pragma unroll
      for (int r = 0; r < 8; ++r) {
        const int row = crow + r;
        float v0 = (ccol0 <= row) ? s0[r] : 0.f;
        float v1 = (ccol1 <= row) ? s1[r] : 0.f;
        A2m[row * 64 + ccol0] = (half_t)v0;
        A2m[row * 64 + ccol1] = (half_t)v1;
        float part = v0 + v1;                       // reduce across 16 lanes
        part += __shfl_xor(part, 1, 32);
        part += __shfl_xor(part, 2, 32);
        part += __shfl_xor(part, 4, 32);
        part += __shfl_xor(part, 8, 32);
        if ((lane & 15) == 0) atomicAdd(&den[row], part);
      }
    }
    __syncthreads();
    // den += Qc . z_prev ; clamp
    if (tid < 64) {
      float a = 0.f;
      const half_t* qp = Qh + (size_t)(t0 + tid) * D_HEAD;
      for (int d2 = 0; d2 < D_HEAD; ++d2) a += (float)qp[d2] * zsh[d2];
      den[tid] = fmaxf(den[tid] + a, 1e-6f);
    }
    __syncthreads();
    // num += tril(QK^T) @ Vc   (B^T = Vt, dv-major)
    {
#pragma unroll
      for (int k = 0; k < 64; k += 32) {
        v16h a  = load_a_frag(A2m + m0 * 64 + k, 64, lane);
        v16h b0 = load_bt_frag(Vth + (size_t)ncp * T_SEQ + t0 + k, T_SEQ, lane);
        v16h b1 = load_bt_frag(Vth + (size_t)(ncp + 16) * T_SEQ + t0 + k, T_SEQ, lane);
        acc0 = wmma_f16(a, b0, acc0);
        acc1 = wmma_f16(a, b1, acc1);
      }
    }
    // write out: Out[t][h*64+dv] = num/den
#pragma unroll
    for (int r = 0; r < 8; ++r) {
      const int row = crow + r;
      const float dd = den[row];
      const size_t og = (size_t)(t0 + row) * (N_HEADS * D_HEAD) + (size_t)h * D_HEAD;
      Out[og + ccol0] = (half_t)(acc0[r] / dd);
      Out[og + ccol1] = (half_t)(acc1[r] / dd);
    }
    // S += Kc^T @ Vc   (A = Kt dh-major, B^T = Vt)
    {
      v8f sc0, sc1;
#pragma unroll
      for (int r = 0; r < 8; ++r) {
        sc0[r] = Ssh[(crow + r) * 64 + ccol0];
        sc1[r] = Ssh[(crow + r) * 64 + ccol1];
      }
      const half_t* Ap = Kth + (size_t)m0 * T_SEQ + t0;
#pragma unroll
      for (int k = 0; k < 64; k += 32) {
        v16h a  = load_a_frag(Ap + k, T_SEQ, lane);
        v16h b0 = load_bt_frag(Vth + (size_t)ncp * T_SEQ + t0 + k, T_SEQ, lane);
        v16h b1 = load_bt_frag(Vth + (size_t)(ncp + 16) * T_SEQ + t0 + k, T_SEQ, lane);
        sc0 = wmma_f16(a, b0, sc0);
        sc1 = wmma_f16(a, b1, sc1);
      }
#pragma unroll
      for (int r = 0; r < 8; ++r) {
        Ssh[(crow + r) * 64 + ccol0] = sc0[r];
        Ssh[(crow + r) * 64 + ccol1] = sc1[r];
      }
    }
    __syncthreads();
    // rebuild S^T f16 for next chunk; z += colsum(Kc)
    for (int i = tid; i < D_HEAD * D_HEAD; i += 256)
      Sth[i] = (half_t)Ssh[(i & 63) * 64 + (i >> 6)];
    if (tid < D_HEAD) {
      float a = zsh[tid];
      const half_t* kp = Kth + (size_t)tid * T_SEQ + t0;
      for (int s2 = 0; s2 < 64; ++s2) a += (float)kp[s2];
      zsh[tid] = a;
    }
    __syncthreads();
  }
}

// ---------------------------------------------------------------------------
// Host-side launcher
// ---------------------------------------------------------------------------
extern "C" void kernel_launch(void* const* d_in, const int* in_sizes, int n_in,
                              void* d_out, int out_size, void* d_ws, size_t ws_size,
                              hipStream_t stream) {
  (void)in_sizes; (void)n_in; (void)out_size; (void)ws_size;

  const float* x     = (const float*)d_in[0];
  const float* ln1_g = (const float*)d_in[1];
  const float* ln1_b = (const float*)d_in[2];
  const float* Wq    = (const float*)d_in[3];
  const float* Wk    = (const float*)d_in[4];
  const float* Wv    = (const float*)d_in[5];
  const float* Wo    = (const float*)d_in[6];
  const float* ln2_g = (const float*)d_in[7];
  const float* ln2_b = (const float*)d_in[8];
  const float* Wf1   = (const float*)d_in[9];
  const float* Wf2   = (const float*)d_in[10];
  float* out = (float*)d_out;

  char* ws = (char*)d_ws;
  const size_t MB = 1024 * 1024;
  half_t* WqT  = (half_t*)(ws + 0 * MB);    // 1024x1024 f16 (2 MB)
  half_t* WkT  = (half_t*)(ws + 2 * MB);
  half_t* WvT  = (half_t*)(ws + 4 * MB);
  half_t* WoT  = (half_t*)(ws + 6 * MB);
  half_t* Wf1T = (half_t*)(ws + 8 * MB);    // 4096x1024 f16 (8 MB)
  half_t* Wf2T = (half_t*)(ws + 16 * MB);   // 1024x4096 f16 (8 MB)
  half_t* Xn   = (half_t*)(ws + 24 * MB);   // 2048x1024 f16 (4 MB)
  half_t* Qn   = (half_t*)(ws + 28 * MB);   // [h][t][dh] (4 MB)
  half_t* KnB  = (half_t*)(ws + 32 * MB);   // [h][t][dh] (4 MB)
  half_t* KtB  = (half_t*)(ws + 36 * MB);   // [h][dh][t] (4 MB)
  half_t* VtB  = (half_t*)(ws + 40 * MB);   // [h][dv][t] (4 MB)
  half_t* Att  = (half_t*)(ws + 44 * MB);   // 2048x1024 f16 (4 MB)
  float*  X1   = (float*)(ws + 48 * MB);    // 2048x1024 f32 (8 MB)
  half_t* H2   = (half_t*)(ws + 56 * MB);   // 2048x1024 f16 (4 MB)
  half_t* G    = (half_t*)(ws + 60 * MB);   // 2048x4096 f16 (16 MB)

  const dim3 blk256(256);
  const dim3 blkg(GEMM_THREADS);
  // weight convert + transpose
  cvt_t_kernel<<<4096,  blk256, 0, stream>>>(Wq,  WqT,  D_MODEL, D_MODEL);
  cvt_t_kernel<<<4096,  blk256, 0, stream>>>(Wk,  WkT,  D_MODEL, D_MODEL);
  cvt_t_kernel<<<4096,  blk256, 0, stream>>>(Wv,  WvT,  D_MODEL, D_MODEL);
  cvt_t_kernel<<<4096,  blk256, 0, stream>>>(Wo,  WoT,  D_MODEL, D_MODEL);
  cvt_t_kernel<<<16384, blk256, 0, stream>>>(Wf1, Wf1T, D_MODEL, D_FF);
  cvt_t_kernel<<<16384, blk256, 0, stream>>>(Wf2, Wf2T, D_FF, D_MODEL);

  // LN1
  ln_kernel<<<T_SEQ, blk256, 0, stream>>>(x, ln1_g, ln1_b, Xn);

  // QKV projections (+phi, layout shuffles); 128x128 staged tiles
  dim3 g_qkv(T_SEQ / BM, D_MODEL / BN);
  gemm_q_phi<<<g_qkv, blkg, 0, stream>>>(Xn, WqT, Qn);
  gemm_k_phi<<<g_qkv, blkg, 0, stream>>>(Xn, WkT, KnB, KtB);
  gemm_v_t  <<<g_qkv, blkg, 0, stream>>>(Xn, WvT, VtB);

  // chunked causal linear attention (one WG per head)
  attn_kernel<<<N_HEADS, blk256, 0, stream>>>(Qn, KnB, KtB, VtB, Att);

  // output projection + residual
  gemm_wo_res<<<dim3(T_SEQ / BM, D_MODEL / BN), blkg, 0, stream>>>(Att, WoT, x, X1);

  // LN2
  ln_kernel<<<T_SEQ, blk256, 0, stream>>>(X1, ln2_g, ln2_b, H2);

  // FFN
  gemm_ffn1_gelu<<<dim3(T_SEQ / BM, D_FF / BN),    blkg, 0, stream>>>(H2, Wf1T, G);
  gemm_ffn2_res <<<dim3(T_SEQ / BM, D_MODEL / BN), blkg, 0, stream>>>(G, Wf2T, X1, out);
}
